// ArceeDecoderLayer_36240934043959
// MI455X (gfx1250) — matrix operand order repaired and checked by reading, not compiled, not run on hardware
//
#include <hip/hip_runtime.h>
#include <hip/hip_bf16.h>

typedef __attribute__((ext_vector_type(16))) __bf16 v16bf;
typedef __attribute__((ext_vector_type(8)))  float  v8f;

#define SEQ    2048
#define HIDDEN 4096
#define NHEADS 32
#define NKV    8
#define HDIM   128
#define INTER  16384
#define QKV_N  6144
#define EPSV   1e-5f
#define THETA  10000.0f

union FragB { v16bf v; unsigned u[8]; };
union FragC { v8f   v; float    f[8]; };

// ---- gfx1250 async global->LDS copy (ASYNCcnt) with compile-safe fallback ----
#if defined(__gfx1250__) && __has_builtin(__builtin_amdgcn_global_load_async_to_lds_b128)
#define HAVE_ASYNC_LDS 1
typedef __attribute__((ext_vector_type(4))) int i4vec;
typedef __attribute__((address_space(1))) i4vec gas_i4;
typedef __attribute__((address_space(3))) i4vec las_i4;
// copy 32 contiguous bytes global->LDS (both 16B aligned)
#define ASYNC_CP32(dst, src)                                                   \
  do {                                                                         \
    gas_i4* g_ = (gas_i4*)(const void*)(src);                                  \
    las_i4* l_ = (las_i4*)(void*)(dst);                                        \
    __builtin_amdgcn_global_load_async_to_lds_b128(g_,     l_,     0, 0);      \
    __builtin_amdgcn_global_load_async_to_lds_b128(g_ + 1, l_ + 1, 0, 0);      \
  } while (0)
// copy 64 contiguous bytes global->LDS
#define ASYNC_CP64(dst, src)                                                   \
  do {                                                                         \
    gas_i4* g_ = (gas_i4*)(const void*)(src);                                  \
    las_i4* l_ = (las_i4*)(void*)(dst);                                        \
    __builtin_amdgcn_global_load_async_to_lds_b128(g_,     l_,     0, 0);      \
    __builtin_amdgcn_global_load_async_to_lds_b128(g_ + 1, l_ + 1, 0, 0);      \
    __builtin_amdgcn_global_load_async_to_lds_b128(g_ + 2, l_ + 2, 0, 0);      \
    __builtin_amdgcn_global_load_async_to_lds_b128(g_ + 3, l_ + 3, 0, 0);      \
  } while (0)
#if __has_builtin(__builtin_amdgcn_s_wait_asynccnt)
#define ASYNC_WAIT() __builtin_amdgcn_s_wait_asynccnt(0)
#else
#define ASYNC_WAIT() asm volatile("s_wait_asynccnt 0x0" ::: "memory")
#endif
#else
#define HAVE_ASYNC_LDS 0
#define ASYNC_CP32(dst, src)                                                   \
  do {                                                                         \
    const uint4* g_ = (const uint4*)(src);                                     \
    uint4* l_ = (uint4*)(dst);                                                 \
    l_[0] = g_[0]; l_[1] = g_[1];                                              \
  } while (0)
#define ASYNC_CP64(dst, src)                                                   \
  do {                                                                         \
    const uint4* g_ = (const uint4*)(src);                                     \
    uint4* l_ = (uint4*)(dst);                                                 \
    l_[0] = g_[0]; l_[1] = g_[1]; l_[2] = g_[2]; l_[3] = g_[3];                \
  } while (0)
#define ASYNC_WAIT() do {} while (0)
#endif

// ---------------------------------------------------------------------------
// Weight convert + transpose: W[K,N] fp32 -> WT[N,K] bf16 (K-contiguous)
// ---------------------------------------------------------------------------
__global__ void wconvert_t(const float* __restrict__ W, __bf16* __restrict__ WT,
                           int K, int N) {
  long idx = (long)blockIdx.x * blockDim.x + threadIdx.x;
  long total = (long)K * N;
  if (idx >= total) return;
  int k = (int)(idx / N);
  int n = (int)(idx - (long)k * N);
  WT[(long)n * K + k] = (__bf16)W[idx];
}

// ---------------------------------------------------------------------------
// Fused (optional add) + RMSNorm; writes fp32 residual (optional) + bf16 h
// ---------------------------------------------------------------------------
__global__ void add_rmsnorm(const float* __restrict__ x, const float* __restrict__ addend,
                            const float* __restrict__ w, float* __restrict__ res_out,
                            __bf16* __restrict__ h_out) {
  int row = blockIdx.x;
  const float* xr = x + (long)row * HIDDEN;
  const float* ar = addend ? addend + (long)row * HIDDEN : nullptr;
  __shared__ float red[8];
  float vals[16];
  float ss = 0.f;
#pragma unroll
  for (int i = 0; i < 16; ++i) {
    int c = threadIdx.x + i * 256;
    float v = xr[c];
    if (ar) v += ar[c];
    vals[i] = v;
    ss += v * v;
  }
#pragma unroll
  for (int m = 16; m > 0; m >>= 1) ss += __shfl_xor(ss, m, 32);
  if ((threadIdx.x & 31) == 0) red[threadIdx.x >> 5] = ss;
  __syncthreads();
  float tot = 0.f;
#pragma unroll
  for (int i = 0; i < 8; ++i) tot += red[i];
  float inv = rsqrtf(tot / (float)HIDDEN + EPSV);
#pragma unroll
  for (int i = 0; i < 16; ++i) {
    int c = threadIdx.x + i * 256;
    float v = vals[i];
    if (res_out) res_out[(long)row * HIDDEN + c] = v;
    h_out[(long)row * HIDDEN + c] = (__bf16)(v * inv * w[c]);
  }
}

// ---------------------------------------------------------------------------
// RoPE-NeoX + split qkv[s][6144] -> q[s][4096], k[s][1024], v[s][1024]
// ---------------------------------------------------------------------------
__global__ void rope_split(const __bf16* __restrict__ qkv, const int* __restrict__ pos,
                           __bf16* __restrict__ qb, __bf16* __restrict__ kb,
                           __bf16* __restrict__ vb) {
  int s = blockIdx.x;
  float p = (float)pos[s];
  const __bf16* src = qkv + (long)s * QKV_N;
  for (int item = threadIdx.x; item < (NHEADS + NKV) * 64; item += 256) {
    int head = item >> 6, j = item & 63;
    float inv_freq = __powf(THETA, -(float)(2 * j) / 128.f);
    float ang = p * inv_freq;
    float sn, cs;
    __sincosf(ang, &sn, &cs);
    if (head < NHEADS) {
      int off = head * 128 + j;
      float x1 = (float)src[off], x2 = (float)src[off + 64];
      qb[(long)s * 4096 + off]      = (__bf16)(x1 * cs - x2 * sn);
      qb[(long)s * 4096 + off + 64] = (__bf16)(x2 * cs + x1 * sn);
    } else {
      int hk = head - NHEADS;
      int off = 4096 + hk * 128 + j;
      float x1 = (float)src[off], x2 = (float)src[off + 64];
      kb[(long)s * 1024 + hk * 128 + j]      = (__bf16)(x1 * cs - x2 * sn);
      kb[(long)s * 1024 + hk * 128 + j + 64] = (__bf16)(x2 * cs + x1 * sn);
    }
  }
  for (int i = threadIdx.x; i < 1024; i += 256)
    vb[(long)s * 1024 + i] = src[5120 + i];
}

// ---------------------------------------------------------------------------
// bf16 WMMA GEMM: C[M,N] = A[M,K] * BT[N,K]^T  (both row-major, K-contiguous)
// 256 threads = 8 waves; BM=BN=128, BK=32; wave -> 32x64 C patch.
// Double-buffered LDS with async global->LDS copies (ASYNCcnt pipeline).
// epilogue: 0=bf16 store, 1=relu^2->bf16, 2=+resid->fp32, 3=fp32 store
// ---------------------------------------------------------------------------
__global__ void __launch_bounds__(256)
gemm_bf16(const __bf16* __restrict__ A, const __bf16* __restrict__ BT,
          void* __restrict__ Cout, const float* __restrict__ resid,
          int M, int N, int K, int epilogue) {
  __shared__ __bf16 As[2][128 * 40];
  __shared__ __bf16 Bs[2][128 * 40];
  int m0 = blockIdx.y * 128, n0 = blockIdx.x * 128;
  int tid = threadIdx.x;
  int lane = tid & 31, wv = tid >> 5;
  int waveM = wv >> 1, waveN = wv & 1;          // 4 x 2 wave grid
  int g = lane >> 4, lr = lane & 15;

  FragC acc[2][4];
#pragma unroll
  for (int mt = 0; mt < 2; ++mt)
#pragma unroll
    for (int nt = 0; nt < 4; ++nt)
#pragma unroll
      for (int r = 0; r < 8; ++r) acc[mt][nt].f[r] = 0.f;

  int ldr = tid >> 1;            // 0..127: tile row
  int ldc = (tid & 1) * 16;      // 0 / 16: 16 bf16 = 32B
  const __bf16* Aptr = A + (long)(m0 + ldr) * K + ldc;
  const __bf16* Bptr = BT + (long)(n0 + ldr) * K + ldc;
  int lofs = ldr * 40 + ldc;

  // prologue: stage tile 0 into buffer 0
  ASYNC_CP32(&As[0][lofs], Aptr);
  ASYNC_CP32(&Bs[0][lofs], Bptr);
  ASYNC_WAIT();
  __syncthreads();

  int cur = 0;
  for (int k0 = 0; k0 < K; k0 += 32) {
    // issue next tile's copies while computing on current
    if (k0 + 32 < K) {
      ASYNC_CP32(&As[cur ^ 1][lofs], Aptr + k0 + 32);
      ASYNC_CP32(&Bs[cur ^ 1][lofs], Bptr + k0 + 32);
#if !HAVE_ASYNC_LDS
    } else {
#else
    }
    if (0) {
#endif
      __builtin_prefetch(Aptr + k0 + 32, 0, 1);
    }

    FragB aF[2], bF[4];
#pragma unroll
    for (int mt = 0; mt < 2; ++mt) {
      const __bf16* base = &As[cur][(waveM * 32 + mt * 16 + lr) * 40];
#pragma unroll
      for (int i = 0; i < 8; ++i) {
        int kk = (i < 4 ? 2 * i : 16 + 2 * (i - 4)) + 8 * g;   // ISA A 16x32 layout
        aF[mt].u[i] = *(const unsigned*)(base + kk);
      }
    }
#pragma unroll
    for (int nt = 0; nt < 4; ++nt) {
      const __bf16* base = &Bs[cur][(waveN * 64 + nt * 16 + lr) * 40];
#pragma unroll
      for (int i = 0; i < 8; ++i)
        bF[nt].u[i] = *(const unsigned*)(base + (i < 4 ? 2 * i : 16 + 2 * (i - 4)) + 8 * g); // ISA B 32x16 layout (same k order as A)
    }
#pragma unroll
    for (int mt = 0; mt < 2; ++mt)
#pragma unroll
      for (int nt = 0; nt < 4; ++nt)
        acc[mt][nt].v = __builtin_amdgcn_wmma_f32_16x16x32_bf16(
            false, aF[mt].v, false, bF[nt].v, (short)0, acc[mt][nt].v, false, false);

    ASYNC_WAIT();
    __syncthreads();
    cur ^= 1;
  }

  // epilogue: C element (VGPR r, lane) -> M = r + 8*(lane>=16), N = lane&15
#pragma unroll
  for (int mt = 0; mt < 2; ++mt)
#pragma unroll
    for (int nt = 0; nt < 4; ++nt)
#pragma unroll
      for (int r = 0; r < 8; ++r) {
        int gm = m0 + waveM * 32 + mt * 16 + r + 8 * g;
        int gn = n0 + waveN * 64 + nt * 16 + lr;
        long idx = (long)gm * N + gn;
        float v = acc[mt][nt].f[r];
        if (epilogue == 0) {
          ((__bf16*)Cout)[idx] = (__bf16)v;
        } else if (epilogue == 1) {
          float u = v > 0.f ? v : 0.f;
          ((__bf16*)Cout)[idx] = (__bf16)(u * u);
        } else if (epilogue == 2) {
          ((float*)Cout)[idx] = v + resid[idx];
        } else {
          ((float*)Cout)[idx] = v;
        }
      }
}

// ---------------------------------------------------------------------------
// Flash attention (causal, GQA rep=4) with WMMA.
// Block = 4 waves = (head, 64 q rows); wave owns 16 q rows.
// K tile 32x128 (K-major, async-staged), V^T tile 128x32, P per-wave LDS.
// ---------------------------------------------------------------------------
__global__ void __launch_bounds__(128)
flash_attn(const __bf16* __restrict__ Q, const __bf16* __restrict__ Kb,
           const __bf16* __restrict__ Vb, __bf16* __restrict__ Ob) {
  __shared__ __bf16 Ks[32 * 136];
  __shared__ __bf16 VT[128 * 36];
  __shared__ __bf16 Ps[4][16 * 40];
  int h = blockIdx.y, qb = blockIdx.x;
  int tid = threadIdx.x, lane = tid & 31, wv = tid >> 5;
  int kvh = h >> 2;
  int q0 = qb * 64 + wv * 16;
  int g = lane >> 4, lr = lane & 15;

  // Q fragments (4 chunks of K=32 over head_dim=128), loaded straight from HBM
  FragB qF[4];
#pragma unroll
  for (int ch = 0; ch < 4; ++ch) {
    const __bf16* base = Q + (long)(q0 + lr) * 4096 + h * 128 + ch * 32;
#pragma unroll
    for (int i = 0; i < 8; ++i) {
      int kk = (i < 4 ? 2 * i : 16 + 2 * (i - 4)) + 8 * g;
      qF[ch].u[i] = *(const unsigned*)(base + kk);
    }
  }

  FragC acc[8];
#pragma unroll
  for (int t = 0; t < 8; ++t)
#pragma unroll
    for (int r = 0; r < 8; ++r) acc[t].f[r] = 0.f;
  float mi[8], li[8];
#pragma unroll
  for (int r = 0; r < 8; ++r) { mi[r] = -3.0e38f; li[r] = 0.f; }
  const float scale = 0.08838834764831845f;   // 1/sqrt(128)

  int krow = tid >> 2, kqu = (tid & 3) * 32;  // 64B (32 el) per lane of K row
  int vkey = tid & 31, vdb = (tid >> 5) * 32;

  int nkb = 2 * qb + 2;                        // causal: keys up to (qb+1)*64
  for (int kb = 0; kb < nkb; ++kb) {
    int kstart = kb * 32;
    __syncthreads();
    // stage K tile [32 keys][128 d], stride 136 el (272B = 17*16, aligned)
    ASYNC_CP64(Ks + krow * 136 + kqu,
               Kb + (long)(kstart + krow) * 1024 + kvh * 128 + kqu);
    // stage V^T tile [128 d][32 keys], stride 36 (transpose via ds scatter)
    {
      const __bf16* src = Vb + (long)(kstart + vkey) * 1024 + kvh * 128 + vdb;
#pragma unroll
      for (int j = 0; j < 32; ++j) VT[(vdb + j) * 36 + vkey] = src[j];
    }
    ASYNC_WAIT();
    __syncthreads();

    // S = Q K^T : two 16x16 score tiles over 32 keys
    FragC s[2];
#pragma unroll
    for (int f = 0; f < 2; ++f)
#pragma unroll
      for (int r = 0; r < 8; ++r) s[f].f[r] = 0.f;
#pragma unroll
    for (int ch = 0; ch < 4; ++ch) {
#pragma unroll
      for (int f = 0; f < 2; ++f) {
        FragB bF;
        const __bf16* base = Ks + (f * 16 + lr) * 136 + ch * 32;
#pragma unroll
        for (int i = 0; i < 8; ++i)
          bF.u[i] = *(const unsigned*)(base + (i < 4 ? 2 * i : 16 + 2 * (i - 4)) + 8 * g);
        s[f].v = __builtin_amdgcn_wmma_f32_16x16x32_bf16(
            false, qF[ch].v, false, bF.v, (short)0, s[f].v, false, false);
      }
    }

    // online softmax; rows live in 16-lane halves -> width-16 reductions
    float p0a[8], p1a[8];
#pragma unroll
    for (int r = 0; r < 8; ++r) {
      int qrow = q0 + r + 8 * g;
      float v0 = s[0].f[r] * scale, v1 = s[1].f[r] * scale;
      if (kstart + lr > qrow)      v0 = -3.0e38f;
      if (kstart + 16 + lr > qrow) v1 = -3.0e38f;
      float mx = fmaxf(v0, v1);
#pragma unroll
      for (int m = 8; m > 0; m >>= 1) mx = fmaxf(mx, __shfl_xor(mx, m, 16));
      float mnew = fmaxf(mi[r], mx);
      float sc = __expf(mi[r] - mnew);
      float p0 = __expf(v0 - mnew), p1 = __expf(v1 - mnew);
      float rs = p0 + p1;
#pragma unroll
      for (int m = 8; m > 0; m >>= 1) rs += __shfl_xor(rs, m, 16);
      li[r] = li[r] * sc + rs;
      mi[r] = mnew;
#pragma unroll
      for (int t = 0; t < 8; ++t) acc[t].f[r] *= sc;
      p0a[r] = p0; p1a[r] = p1;
    }
    // C-layout -> A-layout reshape of P through per-wave LDS
#pragma unroll
    for (int r = 0; r < 8; ++r) {
      int prow = r + 8 * g;
      Ps[wv][prow * 40 + lr]      = (__bf16)p0a[r];
      Ps[wv][prow * 40 + 16 + lr] = (__bf16)p1a[r];
    }
    __asm__ volatile("" ::: "memory");  // same-wave LDS RAW; DS pipe is in-order per wave
    FragB pF;
    {
      const __bf16* base = &Ps[wv][lr * 40];
#pragma unroll
      for (int i = 0; i < 8; ++i) {
        int kk = (i < 4 ? 2 * i : 16 + 2 * (i - 4)) + 8 * g;
        pF.u[i] = *(const unsigned*)(base + kk);
      }
    }
    // O += P V : 8 head-dim tiles of 16
#pragma unroll
    for (int t = 0; t < 8; ++t) {
      FragB vF;
      const __bf16* base = VT + (t * 16 + lr) * 36;
#pragma unroll
      for (int i = 0; i < 8; ++i)
        vF.u[i] = *(const unsigned*)(base + (i < 4 ? 2 * i : 16 + 2 * (i - 4)) + 8 * g);
      acc[t].v = __builtin_amdgcn_wmma_f32_16x16x32_bf16(
          false, pF.v, false, vF.v, (short)0, acc[t].v, false, false);
    }
  }

#pragma unroll
  for (int t = 0; t < 8; ++t)
#pragma unroll
    for (int r = 0; r < 8; ++r) {
      int qrow = q0 + r + 8 * g;
      int d = t * 16 + lr;
      Ob[(long)qrow * 4096 + h * 128 + d] = (__bf16)(acc[t].f[r] / li[r]);
    }
}

// ---------------------------------------------------------------------------
extern "C" void kernel_launch(void* const* d_in, const int* in_sizes, int n_in,
                              void* d_out, int out_size, void* d_ws, size_t ws_size,
                              hipStream_t stream) {
  (void)in_sizes; (void)n_in; (void)out_size; (void)ws_size;
  const int*   positions = (const int*)d_in[0];
  const float* hidden    = (const float*)d_in[1];
  const float* residual  = (const float*)d_in[2];
  const float* qkv_w     = (const float*)d_in[3];
  const float* o_w       = (const float*)d_in[4];
  const float* up_w      = (const float*)d_in[5];
  const float* down_w    = (const float*)d_in[6];
  const float* ln1       = (const float*)d_in[7];
  const float* ln2       = (const float*)d_in[8];

  float* out_mlp  = (float*)d_out;                       // mlp_out [S,H]
  float* out_res2 = out_mlp + (size_t)SEQ * HIDDEN;      // residual2 [S,H]

  char* ws = (char*)d_ws;
  size_t off = 0;
  auto alloc = [&](size_t b) { void* p = ws + off; off += (b + 255) & ~(size_t)255; return p; };
  __bf16* Wt     = (__bf16*)alloc((size_t)INTER * HIDDEN * 2);  // reused for all 4 weights
  float*  resid1 = (float*) alloc((size_t)SEQ * HIDDEN * 4);
  __bf16* hbuf   = (__bf16*)alloc((size_t)SEQ * HIDDEN * 2);    // h1 then h2
  __bf16* qkvo   = (__bf16*)alloc((size_t)SEQ * QKV_N * 2);     // qkv, then attn-out
  __bf16* qbuf   = (__bf16*)alloc((size_t)SEQ * 4096 * 2);
  __bf16* kbuf   = (__bf16*)alloc((size_t)SEQ * 1024 * 2);
  __bf16* vbuf   = (__bf16*)alloc((size_t)SEQ * 1024 * 2);
  __bf16* u2     = (__bf16*)alloc((size_t)SEQ * INTER * 2);

  // 1. residual = hidden + residual; h1 = rmsnorm(residual)*ln1 (bf16)
  add_rmsnorm<<<SEQ, 256, 0, stream>>>(hidden, residual, ln1, resid1, hbuf);

  // 2. qkv = h1 @ qkv_w
  {
    long tot = (long)HIDDEN * QKV_N;
    wconvert_t<<<(unsigned)((tot + 255) / 256), 256, 0, stream>>>(qkv_w, Wt, HIDDEN, QKV_N);
    gemm_bf16<<<dim3(QKV_N / 128, SEQ / 128), 256, 0, stream>>>(
        hbuf, Wt, qkvo, nullptr, SEQ, QKV_N, HIDDEN, 0);
  }

  // 3. RoPE + split
  rope_split<<<SEQ, 256, 0, stream>>>(qkvo, positions, qbuf, kbuf, vbuf);

  // 4. causal GQA attention -> attn [S,4096] (overwrites qkv region)
  flash_attn<<<dim3(SEQ / 64, NHEADS), 128, 0, stream>>>(qbuf, kbuf, vbuf, qkvo);

  // 5. residual2 = attn @ o_w + residual (fp32, second output)
  {
    long tot = (long)HIDDEN * HIDDEN;
    wconvert_t<<<(unsigned)((tot + 255) / 256), 256, 0, stream>>>(o_w, Wt, HIDDEN, HIDDEN);
    gemm_bf16<<<dim3(HIDDEN / 128, SEQ / 128), 256, 0, stream>>>(
        qkvo, Wt, out_res2, resid1, SEQ, HIDDEN, HIDDEN, 2);
  }

  // 6. h2 = rmsnorm(residual2)*ln2
  add_rmsnorm<<<SEQ, 256, 0, stream>>>(out_res2, nullptr, ln2, nullptr, hbuf);

  // 7. u2 = relu(h2 @ up_w)^2
  {
    long tot = (long)HIDDEN * INTER;
    wconvert_t<<<(unsigned)((tot + 255) / 256), 256, 0, stream>>>(up_w, Wt, HIDDEN, INTER);
    gemm_bf16<<<dim3(INTER / 128, SEQ / 128), 256, 0, stream>>>(
        hbuf, Wt, u2, nullptr, SEQ, INTER, HIDDEN, 1);
  }

  // 8. mlp_out = u2 @ down_w (fp32, first output)
  {
    long tot = (long)INTER * HIDDEN;
    wconvert_t<<<(unsigned)((tot + 255) / 256), 256, 0, stream>>>(down_w, Wt, INTER, HIDDEN);
    gemm_bf16<<<dim3(HIDDEN / 128, SEQ / 128), 256, 0, stream>>>(
        u2, Wt, out_mlp, nullptr, SEQ, HIDDEN, INTER, 3);
  }
}
